// T_Transducer_37297495998723
// MI455X (gfx1250) — matrix operand order repaired and checked by reading, not compiled, not run on hardware
//
#include <hip/hip_runtime.h>
#include <stdint.h>

// ---------------------------------------------------------------------------
// Problem constants (match reference)
// ---------------------------------------------------------------------------
#define BB    4
#define TT    150
#define UU    30          // targets length; decoder has UU+1 = 31 positions
#define DD    256
#define NN    2048        // INNER
#define VV    4232
#define NTIL  265         // ceil(VV/16) -> padded V = 4240
#define KK2   64          // NN/32 wmma K-steps for the joint GEMM
#define KK1   8           // DD/32 wmma K-steps for the projection GEMM

#define M_ENC      (BB * TT)         // 600
#define M_ENC_PAD  608
#define M_DEC      (BB * (UU + 1))   // 124
#define M_DEC_PAD  128

typedef __attribute__((ext_vector_type(16))) __bf16 v16bf;
typedef __attribute__((ext_vector_type(8)))  float  v8f;

union Pk {
    uint32_t u[8];
    uint4    q[2];
    v16bf    v;
};

// round-to-nearest-even f32 -> bf16 bits
__device__ __forceinline__ uint32_t f2bf(float f) {
    uint32_t x = __float_as_uint(f);
    return (x + 0x7fffu + ((x >> 16) & 1u)) >> 16;
}
__device__ __forceinline__ uint32_t pack_bf2(float lo, float hi) {
    return f2bf(lo) | (f2bf(hi) << 16);
}

// A-operand K offset for 16x32 bf16 A tile (ISA 7.12.2):
// lanes 0-15 hold K {0..7, 16..23}; lanes 16-31 hold K {8..15, 24..31}
__device__ __forceinline__ int koffA(int j, int hi) {
    return ((j < 4) ? 0 : 16) + hi * 8 + (j & 3) * 2;
}
// B-operand K offset for 32x16 bf16 B tile: lanes 0-15 K {0..7, 16..23}; lanes 16-31 K {8..15, 24..31}
__device__ __forceinline__ int koffB(int j, int hi) {
    return ((j < 4) ? 0 : 16) + hi * 8 + (j & 3) * 2;
}

// ---------------------------------------------------------------------------
// Kernel 0: repack W2 (NN x VV f32) into bf16 WMMA-B lane order.
// Packed index: ((nt*64 + kk)*32 + lane)*8 + j   (uint32 holding 2 bf16)
// ---------------------------------------------------------------------------
__global__ void pack_w2(const float* __restrict__ W2, uint32_t* __restrict__ P) {
    long idx = (long)blockIdx.x * 256 + threadIdx.x;
    const long total = (long)NTIL * KK2 * 32 * 8;
    if (idx >= total) return;
    int j    = (int)(idx & 7);
    int lane = (int)((idx >> 3) & 31);
    int kk   = (int)((idx >> 8) & 63);
    int nt   = (int)(idx >> 14);
    int lo = lane & 15, hi = lane >> 4;
    int col = nt * 16 + lo;
    int k0  = kk * 32 + koffB(j, hi);
    float b0 = (col < VV) ? W2[(size_t)k0 * VV + col]       : 0.0f;
    float b1 = (col < VV) ? W2[(size_t)(k0 + 1) * VV + col] : 0.0f;
    P[idx] = pack_bf2(b0, b1);
}

// ---------------------------------------------------------------------------
// Kernel 1: projection GEMM  Y(Mpad x 2048) = X(M x 256) * W(256 x 2048).
// One wave per 16x16 output tile, bf16 WMMA over K=256. Output buffer is
// padded to a 16-row multiple so stores are unconditional (no exec juggling).
// ---------------------------------------------------------------------------
__global__ void proj_gemm(const float* __restrict__ X, const float* __restrict__ W,
                          float* __restrict__ Y, int M) {
    int lane = threadIdx.x & 31;
    int lo = lane & 15, hi = lane >> 4;
    int mt = blockIdx.y, nt = blockIdx.x;
    int row = mt * 16 + lo;
    int rclamp = (row < M) ? row : (M - 1);
    int col = nt * 16 + lo;

    v8f c;
#pragma unroll
    for (int i = 0; i < 8; ++i) c[i] = 0.0f;

    for (int kk = 0; kk < KK1; ++kk) {
        int kbase = kk * 32;
        Pk a, b;
#pragma unroll
        for (int j = 0; j < 8; ++j) {
            int ka = kbase + koffA(j, hi);
            float a0 = X[(size_t)rclamp * DD + ka];
            float a1 = X[(size_t)rclamp * DD + ka + 1];
            a.u[j] = pack_bf2(a0, a1);
            int kb = kbase + koffB(j, hi);
            float b0 = W[(size_t)kb * NN + col];
            float b1 = W[(size_t)(kb + 1) * NN + col];
            b.u[j] = pack_bf2(b0, b1);
        }
        c = __builtin_amdgcn_wmma_f32_16x16x32_bf16(false, a.v, false, b.v,
                                                    (short)0, c, false, false);
    }
#pragma unroll
    for (int j = 0; j < 8; ++j) {
        int r = mt * 16 + hi * 8 + j;      // C/D layout: row = hi*8+j, col = lo
        Y[(size_t)r * NN + col] = c[j];    // padded buffer: always in-bounds
    }
}

// ---------------------------------------------------------------------------
// Kernel 2: fused joint-net + online log-softmax reduction.
// Grid: (B*T). Block: 256 threads = 8 waves. One WG covers ALL 32 decoder
// rows (2 M-tiles) so every W2 B-tile load feeds two WMMAs (halves L2 reads).
// Phase 1: A = tanh(h_enc + h_dec + b1) -> LDS (bf16, WMMA-A lane order),
//          32 rows x 2048 K = 128 KB.
// Phase 2: each wave strides N-tiles, WMMA K-loop x2 M-tiles, online
//          logsumexp; blank/emit logits captured straight to LDS.
// ---------------------------------------------------------------------------
__global__ void joint_kernel(const float* __restrict__ h_enc,
                             const float* __restrict__ h_dec,
                             const float* __restrict__ b1,
                             const uint32_t* __restrict__ w2p,
                             const float* __restrict__ b2,
                             const int* __restrict__ targets,
                             float* __restrict__ blank_lp,   // (B*T, 31)
                             float* __restrict__ emit_lp) {  // (B*T, 30)
    extern __shared__ uint32_t ldsA[];        // 2 * 64*32*8 u32 = 128 KB
    __shared__ float waveM[8][32], waveS[8][32];
    __shared__ float ldsB[32], ldsE[32];

    const int bt = blockIdx.x;                // 0..B*T-1
    const int b  = bt / TT;
    const int tid  = threadIdx.x;
    const int wave = tid >> 5, lane = tid & 31;
    const int lo = lane & 15, hi = lane >> 4;

    // ---- Phase 1: activations into LDS in WMMA-A order -------------------
    const float* he = h_enc + (size_t)bt * NN;
    for (int idx = tid; idx < 32 * NN; idx += 256) {
        int u = idx >> 11;                    // decoder row 0..31 (31 = pad)
        int k = idx & (NN - 1);
        float hd = (u < UU + 1) ? h_dec[(size_t)(b * (UU + 1) + u) * NN + k] : 0.0f;
        float a  = tanhf(he[k] + hd + b1[k]);
        int mt = u >> 4, rowl = u & 15;
        int kk = k >> 5, rem = k & 31;
        int p = rem >> 1, lowbit = rem & 1;   // invert koffA
        int hi2 = (p >> 2) & 1;
        int j   = (p & 3) | ((p >> 3) << 2);
        uint16_t* dst = (uint16_t*)&ldsA[mt * 16384 + (kk * 32 + hi2 * 16 + rowl) * 8 + j];
        dst[lowbit] = (uint16_t)f2bf(a);
    }
    __syncthreads();

    // ---- Phase 2: per-wave N-tile loop, 2 M-tiles per B load -------------
    float mx[2][8], sm[2][8];
    int   ecol[2][8];
#pragma unroll
    for (int mt = 0; mt < 2; ++mt)
#pragma unroll
        for (int j = 0; j < 8; ++j) {
            mx[mt][j] = -__builtin_inff();
            sm[mt][j] = 0.0f;
            int u = mt * 16 + hi * 8 + j;
            ecol[mt][j] = (u < UU) ? targets[b * UU + u] : -1;
        }

    const uint4* aBase = (const uint4*)ldsA;
    for (int nt = wave; nt < NTIL; nt += 8) {
        v8f c0, c1;
#pragma unroll
        for (int i = 0; i < 8; ++i) { c0[i] = 0.0f; c1[i] = 0.0f; }

        const uint4* bp  = (const uint4*)w2p + ((size_t)nt * 2048 + lane) * 2;
        if (nt + 8 < NTIL)   // pull next tile toward L2 (global_prefetch_b8)
            __builtin_prefetch((const uint4*)w2p + ((size_t)(nt + 8) * 2048 + lane) * 2, 0, 1);

        const uint4* ap0 = aBase + lane * 2;
        const uint4* ap1 = ap0 + 4096;        // second M-tile, +16384 u32
#pragma unroll 4
        for (int kk = 0; kk < KK2; ++kk) {
            Pk a0, a1, bq;
            bq.q[0] = bp[0];  bq.q[1] = bp[1];
            a0.q[0] = ap0[0]; a0.q[1] = ap0[1];
            a1.q[0] = ap1[0]; a1.q[1] = ap1[1];
            ap0 += 64; ap1 += 64; bp += 64;
            c0 = __builtin_amdgcn_wmma_f32_16x16x32_bf16(false, a0.v, false, bq.v,
                                                         (short)0, c0, false, false);
            c1 = __builtin_amdgcn_wmma_f32_16x16x32_bf16(false, a1.v, false, bq.v,
                                                         (short)0, c1, false, false);
        }

        int col = nt * 16 + lo;
        if (col < VV) {
            float bias = b2[col];
#pragma unroll
            for (int mt = 0; mt < 2; ++mt) {
#pragma unroll
                for (int j = 0; j < 8; ++j) {
                    float v = (mt ? c1[j] : c0[j]) + bias;
                    if (col == 0)            ldsB[mt * 16 + hi * 8 + j] = v;
                    if (col == ecol[mt][j])  ldsE[mt * 16 + hi * 8 + j] = v;
                    float m2 = fmaxf(mx[mt][j], v);
                    sm[mt][j] = sm[mt][j] * __expf(mx[mt][j] - m2) + __expf(v - m2);
                    mx[mt][j] = m2;
                }
            }
        }
    }

    // cross-lane logsumexp combine within each 16-lane half
#pragma unroll
    for (int m = 1; m < 16; m <<= 1) {
#pragma unroll
        for (int mt = 0; mt < 2; ++mt)
#pragma unroll
            for (int j = 0; j < 8; ++j) {
                float om = __shfl_xor(mx[mt][j], m, 32);
                float os = __shfl_xor(sm[mt][j], m, 32);
                float mm = fmaxf(mx[mt][j], om);
                sm[mt][j] = sm[mt][j] * __expf(mx[mt][j] - mm) + os * __expf(om - mm);
                mx[mt][j] = mm;
            }
    }
    if (lo == 0) {
#pragma unroll
        for (int mt = 0; mt < 2; ++mt)
#pragma unroll
            for (int j = 0; j < 8; ++j) {
                waveM[wave][mt * 16 + hi * 8 + j] = mx[mt][j];
                waveS[wave][mt * 16 + hi * 8 + j] = sm[mt][j];
            }
    }
    __syncthreads();

    // final cross-wave combine + output
    if (tid < 32) {
        float M = waveM[0][tid], S = waveS[0][tid];
#pragma unroll
        for (int w = 1; w < 8; ++w) {
            float om = waveM[w][tid], os = waveS[w][tid];
            float mm = fmaxf(M, om);
            S = S * __expf(M - mm) + os * __expf(om - mm);
            M = mm;
        }
        float logZ = M + __logf(S);
        int u = tid;
        if (u < UU + 1) blank_lp[(size_t)bt * (UU + 1) + u] = ldsB[u] - logZ;
        if (u < UU)     emit_lp [(size_t)bt * UU + u]       = ldsE[u] - logZ;
    }
}

// ---------------------------------------------------------------------------
// Kernel 3: RNN-T alpha lattice + mean NLL (tiny serial tail).
// ---------------------------------------------------------------------------
__global__ void alpha_kernel(const float* __restrict__ blank_lp,
                             const float* __restrict__ emit_lp,
                             const int* __restrict__ in_len,
                             const int* __restrict__ tg_len,
                             float* __restrict__ out) {
    int b = threadIdx.x;
    float nll = 0.0f;
    if (b < BB) {
        const float* blp = blank_lp + (size_t)b * TT * (UU + 1);
        const float* elp = emit_lp  + (size_t)b * TT * UU;
        float a[UU + 1];
        a[0] = 0.0f;
        for (int u = 1; u <= UU; ++u) a[u] = a[u - 1] + elp[u - 1];
        int ti = in_len[b] - 1, ui = tg_len[b];
        float ll_a = (ti == 0) ? a[ui] : 0.0f;
        for (int t = 1; t < TT; ++t) {
            float fb[UU + 1];
            for (int u = 0; u <= UU; ++u) fb[u] = a[u] + blp[(t - 1) * (UU + 1) + u];
            a[0] = fb[0];
            for (int u = 1; u <= UU; ++u) {
                float x = fb[u], y = a[u - 1] + elp[t * UU + (u - 1)];
                float m = fmaxf(x, y);
                a[u] = m + __logf(__expf(x - m) + __expf(y - m));
            }
            if (t == ti) ll_a = a[ui];
        }
        nll = -(ll_a + blp[ti * (UU + 1) + ui]);
    }
    nll += __shfl_xor(nll, 1, 32);
    nll += __shfl_xor(nll, 2, 32);
    if (threadIdx.x == 0) out[0] = nll / (float)BB;
}

// ---------------------------------------------------------------------------
// Launch
// ---------------------------------------------------------------------------
extern "C" void kernel_launch(void* const* d_in, const int* in_sizes, int n_in,
                              void* d_out, int out_size, void* d_ws, size_t ws_size,
                              hipStream_t stream) {
    (void)in_sizes; (void)n_in; (void)out_size; (void)ws_size;
    const float* enc  = (const float*)d_in[0];
    const float* dec  = (const float*)d_in[1];
    const float* Wenc = (const float*)d_in[2];
    const float* Wdec = (const float*)d_in[3];
    const float* b1   = (const float*)d_in[4];
    const float* W2   = (const float*)d_in[5];
    const float* b2   = (const float*)d_in[6];
    const int* targets = (const int*)d_in[7];
    const int* in_len  = (const int*)d_in[8];
    const int* tg_len  = (const int*)d_in[9];

    char* ws = (char*)d_ws;
    const size_t w2p_bytes   = (size_t)NTIL * KK2 * 32 * 8 * 4;   // 17,367,040
    const size_t henc_bytes  = (size_t)M_ENC_PAD * NN * 4;        //  4,980,736
    const size_t hdec_bytes  = (size_t)M_DEC_PAD * NN * 4;        //  1,048,576
    const size_t blank_bytes = (size_t)BB * TT * (UU + 1) * 4;    //     74,400
    uint32_t* w2p   = (uint32_t*)ws;
    float* h_enc    = (float*)(ws + w2p_bytes);
    float* h_dec    = (float*)(ws + w2p_bytes + henc_bytes);
    float* blank_lp = (float*)(ws + w2p_bytes + henc_bytes + hdec_bytes);
    float* emit_lp  = (float*)(ws + w2p_bytes + henc_bytes + hdec_bytes + blank_bytes);

    // 0) repack W2 to bf16 WMMA-B layout (resident in L2 thereafter)
    const long packN = (long)NTIL * KK2 * 32 * 8;
    pack_w2<<<(unsigned)((packN + 255) / 256), 256, 0, stream>>>(W2, w2p);

    // 1) projections: h_enc (600->608 x 2048), h_dec (124->128 x 2048)
    proj_gemm<<<dim3(NN / 16, M_ENC_PAD / 16), 32, 0, stream>>>(enc, Wenc, h_enc, M_ENC);
    proj_gemm<<<dim3(NN / 16, M_DEC_PAD / 16), 32, 0, stream>>>(dec, Wdec, h_dec, M_DEC);

    // 2) fused joint-net GEMM + online log-softmax (dominant: ~322 GFLOP bf16 WMMA)
    joint_kernel<<<dim3(BB * TT), 256, 131072, stream>>>(
        h_enc, h_dec, b1, w2p, b2, targets, blank_lp, emit_lp);

    // 3) alpha lattice + loss
    alpha_kernel<<<1, 32, 0, stream>>>(blank_lp, emit_lp, in_len, tg_len, (float*)d_out);
}
